// selfAttentionV1_64544768524377
// MI455X (gfx1250) — matrix operand order repaired and checked by reading, not compiled, not run on hardware
//
#include <hip/hip_runtime.h>
#include <hip/hip_bf16.h>
#include <math.h>

typedef __bf16 bf16_t;
typedef __attribute__((ext_vector_type(16))) __bf16 v16bf;
typedef __attribute__((ext_vector_type(8)))  __bf16 v8bf;
typedef __attribute__((ext_vector_type(4)))  __bf16 v4bf;
typedef __attribute__((ext_vector_type(8)))  float  v8f;
typedef __attribute__((ext_vector_type(4)))  float  v4f;

#define B_ 4
#define S_ 2048
#define E_ 1024
#define M_ (B_ * S_)   /* 8192 */
#define F_ (3 * E_)    /* 3072 */
#define LDST 72        /* LDS row stride (bf16) for 64-deep K tiles: 144B = 36 banks */

static __device__ __forceinline__ v8f wmma_bf16(v16bf a, v16bf b, v8f c) {
    return __builtin_amdgcn_wmma_f32_16x16x32_bf16(
        false, a, false, b, (short)0, c, false, false);
}

static __device__ __forceinline__ v16bf combine8(v8bf lo, v8bf hi) {
    v16bf r;
#pragma unroll
    for (int i = 0; i < 8; ++i) { r[i] = lo[i]; r[8 + i] = hi[i]; }
    return r;
}

// A fragment from f32 row: elems 0-7 -> K=k0+8h.., 8-15 -> K=k0+16+8h..
static __device__ __forceinline__ v16bf a_from_f32(const float* row, int k0, int half) {
    v4f a0 = *(const v4f*)(row + k0 + 8 * half);
    v4f a1 = *(const v4f*)(row + k0 + 8 * half + 4);
    v4f a2 = *(const v4f*)(row + k0 + 16 + 8 * half);
    v4f a3 = *(const v4f*)(row + k0 + 16 + 8 * half + 4);
    v16bf a;
#pragma unroll
    for (int i = 0; i < 4; ++i) {
        a[i]      = (bf16_t)a0[i];
        a[4 + i]  = (bf16_t)a1[i];
        a[8 + i]  = (bf16_t)a2[i];
        a[12 + i] = (bf16_t)a3[i];
    }
    return a;
}

static __device__ __forceinline__ v16bf a_from_bf16(const bf16_t* row, int k0, int half) {
    v8bf lo = *(const v8bf*)(row + k0 + 8 * half);
    v8bf hi = *(const v8bf*)(row + k0 + 16 + 8 * half);
    return combine8(lo, hi);
}

// B fragment from staged 64x64 LDS tile; koff in {0,32} selects K sub-tile
static __device__ __forceinline__ v16bf b_from_lds(const bf16_t* tile, int ln, int half,
                                                   int t, int koff) {
    const bf16_t* p = tile + (t * 16 + ln) * LDST + koff + 8 * half;
    v8bf lo = *(const v8bf*)p;        // 16B-aligned ds_load_b128
    v8bf hi = *(const v8bf*)(p + 16);
    return combine8(lo, hi);
}

// ---------------------------------------------------------------------------
// Kernel 1: qkv = X @ W^T + b  (f32 in; Q/K/Vt staged bf16, Q pre-scaled)
// Workgroup: 256(M) x 64(N); wave: 32x64. W tile 64x64 staged f32->bf16 in LDS.
// ---------------------------------------------------------------------------
__global__ __launch_bounds__(256) void qkv_kernel(
    const float* __restrict__ X, const float* __restrict__ W,
    const float* __restrict__ bias,
    bf16_t* __restrict__ Q, bf16_t* __restrict__ Km, bf16_t* __restrict__ Vt)
{
    __shared__ __align__(16) bf16_t tile[2][64 * LDST];
    const int tid  = threadIdx.x;
    const int lane = tid & 31, wave = tid >> 5;
    const int half = lane >> 4, ln = lane & 15;
    const int m0 = blockIdx.x * 256 + wave * 32;
    const int f0 = blockIdx.y * 64;

    const float* xrow0 = X + (size_t)(m0 + ln) * E_;
    const float* xrow1 = X + (size_t)(m0 + 16 + ln) * E_;

    // coalesced stage: 64 rows x 64 f32 -> bf16; 4 float4-chunks per thread
    auto stage = [&](int buf, int k0) {
#pragma unroll
        for (int j = 0; j < 4; ++j) {
            const int c   = tid + 256 * j;
            const int row = c >> 4;       // 0..63
            const int c4  = c & 15;       // float4 within 64-wide row
            v4f w4 = *(const v4f*)(W + (size_t)(f0 + row) * E_ + k0 + c4 * 4);
            v4bf o;
            o[0] = (bf16_t)w4[0]; o[1] = (bf16_t)w4[1];
            o[2] = (bf16_t)w4[2]; o[3] = (bf16_t)w4[3];
            *(v4bf*)&tile[buf][row * LDST + c4 * 4] = o;
        }
    };

    v8f acc[2][4] = {};
    auto compute32 = [&](int buf, int k0, int koff) {
        v16bf a0 = a_from_f32(xrow0, k0 + koff, half);
        v16bf a1 = a_from_f32(xrow1, k0 + koff, half);
        v16bf bfr[4];
#pragma unroll
        for (int t = 0; t < 4; ++t) bfr[t] = b_from_lds(&tile[buf][0], ln, half, t, koff);
#pragma unroll
        for (int t = 0; t < 4; ++t) {
            acc[0][t] = wmma_bf16(a0, bfr[t], acc[0][t]);
            acc[1][t] = wmma_bf16(a1, bfr[t], acc[1][t]);
        }
    };

    stage(0, 0);
    __syncthreads();
    for (int k0 = 0; k0 < E_; k0 += 64) {
        const int buf = (k0 >> 6) & 1;
        if (k0 + 64 < E_) stage(buf ^ 1, k0 + 64);   // prefetch next 64-deep tile
        compute32(buf, k0, 0);
        compute32(buf, k0, 32);
        __syncthreads();
    }

    const float qscale = 0.03125f;  // 1/sqrt(E)
#pragma unroll
    for (int rt = 0; rt < 2; ++rt) {
#pragma unroll
        for (int t = 0; t < 4; ++t) {
            const int f = f0 + 16 * t + ln;
            const float bv = bias[f];
#pragma unroll
            for (int r = 0; r < 8; ++r) {
                const int m = m0 + rt * 16 + r + 8 * half;
                const float v = acc[rt][t][r] + bv;
                if (f < E_) {
                    Q[(size_t)m * E_ + f] = (bf16_t)(v * qscale);
                } else if (f < 2 * E_) {
                    Km[(size_t)m * E_ + (f - E_)] = (bf16_t)v;
                } else {
                    const int e = f - 2 * E_;
                    const int bidx = m >> 11;
                    const int s = m & (S_ - 1);
                    Vt[((size_t)bidx * E_ + e) * S_ + s] = (bf16_t)v;
                }
            }
        }
    }
}

// ---------------------------------------------------------------------------
// Kernel 2: scores = Qs @ K^T -> raw f32 into attention region of d_out
// ---------------------------------------------------------------------------
__global__ __launch_bounds__(256) void scores_kernel(
    const bf16_t* __restrict__ Q, const bf16_t* __restrict__ Km,
    float* __restrict__ attn)
{
    __shared__ __align__(16) bf16_t tile[2][64 * LDST];
    const int tid  = threadIdx.x;
    const int lane = tid & 31, wave = tid >> 5;
    const int half = lane >> 4, ln = lane & 15;
    const int b  = blockIdx.z;
    const int q0 = blockIdx.x * 256 + wave * 32;
    const int n0 = blockIdx.y * 64;

    const bf16_t* qrow0 = Q + (size_t)(b * S_ + q0 + ln) * E_;
    const bf16_t* qrow1 = qrow0 + (size_t)16 * E_;

    // coalesced stage: 64 key-rows x 64 bf16; 2 x 16B chunks per thread
    auto stage = [&](int buf, int k0) {
#pragma unroll
        for (int j = 0; j < 2; ++j) {
            const int c   = tid + 256 * j;
            const int row = c >> 3;       // 0..63
            const int ch  = c & 7;        // 8-bf16 chunk within 64-wide row
            v8bf d = *(const v8bf*)(Km + (size_t)(b * S_ + n0 + row) * E_ + k0 + ch * 8);
            *(v8bf*)&tile[buf][row * LDST + ch * 8] = d;
        }
    };

    v8f acc[2][4] = {};
    auto compute32 = [&](int buf, int k0, int koff) {
        v16bf a0 = a_from_bf16(qrow0, k0 + koff, half);
        v16bf a1 = a_from_bf16(qrow1, k0 + koff, half);
        v16bf bfr[4];
#pragma unroll
        for (int t = 0; t < 4; ++t) bfr[t] = b_from_lds(&tile[buf][0], ln, half, t, koff);
#pragma unroll
        for (int t = 0; t < 4; ++t) {
            acc[0][t] = wmma_bf16(a0, bfr[t], acc[0][t]);
            acc[1][t] = wmma_bf16(a1, bfr[t], acc[1][t]);
        }
    };

    stage(0, 0);
    __syncthreads();
    for (int k0 = 0; k0 < E_; k0 += 64) {
        const int buf = (k0 >> 6) & 1;
        if (k0 + 64 < E_) stage(buf ^ 1, k0 + 64);
        compute32(buf, k0, 0);
        compute32(buf, k0, 32);
        __syncthreads();
    }

    float* arow = attn + (size_t)b * S_ * S_;
#pragma unroll
    for (int rt = 0; rt < 2; ++rt) {
#pragma unroll
        for (int t = 0; t < 4; ++t) {
            const int n = n0 + 16 * t + ln;
#pragma unroll
            for (int r = 0; r < 8; ++r) {
                const int q = q0 + rt * 16 + r + 8 * half;
                arow[(size_t)q * S_ + n] = acc[rt][t][r];
            }
        }
    }
}

// ---------------------------------------------------------------------------
// Kernel 3: row softmax in place
// ---------------------------------------------------------------------------
__global__ __launch_bounds__(256) void softmax_kernel(float* __restrict__ attn)
{
    float* p = attn + (size_t)blockIdx.x * S_;
    const int tid = threadIdx.x;

    float v[8];
    float mx = -3.402823466e38f;
#pragma unroll
    for (int j = 0; j < 8; ++j) { v[j] = p[tid + 256 * j]; mx = fmaxf(mx, v[j]); }

    __shared__ float red[256];
    red[tid] = mx;
    __syncthreads();
    for (int s2 = 128; s2 > 0; s2 >>= 1) {
        if (tid < s2) red[tid] = fmaxf(red[tid], red[tid + s2]);
        __syncthreads();
    }
    mx = red[0];
    __syncthreads();

    float sum = 0.f;
#pragma unroll
    for (int j = 0; j < 8; ++j) { v[j] = __expf(v[j] - mx); sum += v[j]; }
    red[tid] = sum;
    __syncthreads();
    for (int s2 = 128; s2 > 0; s2 >>= 1) {
        if (tid < s2) red[tid] += red[tid + s2];
        __syncthreads();
    }
    const float inv = 1.0f / red[0];
#pragma unroll
    for (int j = 0; j < 8; ++j) p[tid + 256 * j] = v[j] * inv;
}

// ---------------------------------------------------------------------------
// Kernel 4: output = P @ V  (P f32 -> bf16 on the fly; V^T tile via LDS)
// ---------------------------------------------------------------------------
__global__ __launch_bounds__(256) void av_kernel(
    const float* __restrict__ attn, const bf16_t* __restrict__ Vt,
    float* __restrict__ out)
{
    __shared__ __align__(16) bf16_t tile[2][64 * LDST];
    const int tid  = threadIdx.x;
    const int lane = tid & 31, wave = tid >> 5;
    const int half = lane >> 4, ln = lane & 15;
    const int b  = blockIdx.z;
    const int q0 = blockIdx.x * 256 + wave * 32;
    const int e0 = blockIdx.y * 64;

    const float* prow0 = attn + (size_t)(b * S_ + q0 + ln) * S_;
    const float* prow1 = prow0 + (size_t)16 * S_;

    auto stage = [&](int buf, int k0) {
#pragma unroll
        for (int j = 0; j < 2; ++j) {
            const int c   = tid + 256 * j;
            const int row = c >> 3;
            const int ch  = c & 7;
            v8bf d = *(const v8bf*)(Vt + ((size_t)b * E_ + e0 + row) * S_ + k0 + ch * 8);
            *(v8bf*)&tile[buf][row * LDST + ch * 8] = d;
        }
    };

    v8f acc[2][4] = {};
    auto compute32 = [&](int buf, int k0, int koff) {
        v16bf a0 = a_from_f32(prow0, k0 + koff, half);
        v16bf a1 = a_from_f32(prow1, k0 + koff, half);
        v16bf bfr[4];
#pragma unroll
        for (int t = 0; t < 4; ++t) bfr[t] = b_from_lds(&tile[buf][0], ln, half, t, koff);
#pragma unroll
        for (int t = 0; t < 4; ++t) {
            acc[0][t] = wmma_bf16(a0, bfr[t], acc[0][t]);
            acc[1][t] = wmma_bf16(a1, bfr[t], acc[1][t]);
        }
    };

    stage(0, 0);
    __syncthreads();
    for (int k0 = 0; k0 < S_; k0 += 64) {
        const int buf = (k0 >> 6) & 1;
        if (k0 + 64 < S_) stage(buf ^ 1, k0 + 64);
        compute32(buf, k0, 0);
        compute32(buf, k0, 32);
        __syncthreads();
    }

#pragma unroll
    for (int rt = 0; rt < 2; ++rt) {
#pragma unroll
        for (int t = 0; t < 4; ++t) {
            const int e = e0 + 16 * t + ln;
#pragma unroll
            for (int r = 0; r < 8; ++r) {
                const int q = q0 + rt * 16 + r + 8 * half;
                out[(size_t)(b * S_ + q) * E_ + e] = acc[rt][t][r];
            }
        }
    }
}

// ---------------------------------------------------------------------------
extern "C" void kernel_launch(void* const* d_in, const int* in_sizes, int n_in,
                              void* d_out, int out_size, void* d_ws, size_t ws_size,
                              hipStream_t stream)
{
    const float* X    = (const float*)d_in[0];
    const float* W    = (const float*)d_in[1];
    const float* bias = (const float*)d_in[2];

    float* out  = (float*)d_out;
    float* attn = out + (size_t)M_ * E_;

    bf16_t* Q  = (bf16_t*)d_ws;
    bf16_t* Km = Q  + (size_t)M_ * E_;
    bf16_t* Vt = Km + (size_t)M_ * E_;

    qkv_kernel    <<<dim3(M_ / 256, F_ / 64),     256, 0, stream>>>(X, W, bias, Q, Km, Vt);
    scores_kernel <<<dim3(S_ / 256, S_ / 64, B_), 256, 0, stream>>>(Q, Km, attn);
    softmax_kernel<<<dim3(M_),                    256, 0, stream>>>(attn);
    av_kernel     <<<dim3(S_ / 256, E_ / 64, B_), 256, 0, stream>>>(attn, Vt, out);
}
